// MultiHeadCrossWindowAttention_67001489818025
// MI455X (gfx1250) — compile-verified, matrix operand order checked by reading, not run on hardware
//
#include <hip/hip_runtime.h>

// ---------------------------------------------------------------------------
// Shifted-window multi-head attention for MI455X (gfx1250), bf16 WMMA path.
// One workgroup (4 wave32) per (batch, head, window). All per-window operands
// staged in LDS as bf16; f32 accumulation via v_wmma_f32_16x16x32_bf16.
// Global staging/scatter fully vectorized as float4 (window rows are 16B-
// aligned contiguous quads, and the +/-4 roll wraps on quad boundaries).
// ---------------------------------------------------------------------------

typedef __attribute__((ext_vector_type(16))) __bf16 v16bf;
typedef __attribute__((ext_vector_type(8)))  float  v8f;

namespace {
constexpr int WSZ   = 8;     // window size
constexpr int HD    = 64;    // head dim
constexpr int DISP  = 4;     // shift
constexpr int IMG   = 128;   // spatial H=W
constexpr int CH    = 512;   // channels = 8 heads * 64
constexpr int NW    = 16;    // windows per spatial dim
constexpr int LDA   = 72;    // bf16 LDS row pitch (64 + 8 pad)
constexpr int LDY   = 68;    // f32  LDS row pitch
}

// K-index inside a 16-element bf16 fragment for a given lane-half.
// Matches ISA 16-bit A layout: VGPRs hold K = {0..7,16..23} (lanes 0-15)
// or {8..15,24..31} (lanes 16-31), two K per dword.
__device__ __forceinline__ int frag_k(int e, int half) {
  int v = e >> 1;
  return ((v < 4) ? 0 : 16) + ((v & 3) << 1) + (half << 3) + (e & 1);
}

// A fragment: rows m0..m0+15 (one per lane mod 16), K = k0..k0+31.
__device__ __forceinline__ v16bf load_a(const __bf16* M, int m0, int k0, int lane) {
  const int half = lane >> 4;
  const int m    = m0 + (lane & 15);
  v16bf a;
#pragma unroll
  for (int e = 0; e < 16; ++e) a[e] = M[m * LDA + k0 + frag_k(e, half)];
  return a;
}

// B fragment with transposed source: B[k][n] = M[n][k]  (for W^T and K^T)
__device__ __forceinline__ v16bf load_bt(const __bf16* M, int n0, int k0, int lane) {
  const int half = lane >> 4;
  const int n    = n0 + (lane & 15);
  v16bf b;
#pragma unroll
  for (int e = 0; e < 16; ++e) b[e] = M[n * LDA + k0 + frag_k(e, half)];
  return b;
}

// B fragment, natural source: B[k][n] = M[k][n]  (for V in P@V)
__device__ __forceinline__ v16bf load_bn(const __bf16* M, int k0, int n0, int lane) {
  const int half = lane >> 4;
  const int n    = n0 + (lane & 15);
  v16bf b;
#pragma unroll
  for (int e = 0; e < 16; ++e) b[e] = M[(k0 + frag_k(e, half)) * LDA + n];
  return b;
}

// Store one 16x16 f32 accumulator tile as bf16 (C layout: M = r + 8*half).
__device__ __forceinline__ void store_c(__bf16* D, int m0, int n0, v8f c, int lane) {
  const int half = lane >> 4;
  const int n    = n0 + (lane & 15);
#pragma unroll
  for (int r = 0; r < 8; ++r)
    D[(m0 + (half << 3) + r) * LDA + n] = (__bf16)c[r];
}

__device__ __forceinline__ void store_strip(__bf16* D, int m0, const v8f acc[4], int lane) {
#pragma unroll
  for (int t = 0; t < 4; ++t) store_c(D, m0, t * 16, acc[t], lane);
}

// 16x64 strip of C = A(16x64) * B(64x64), B given transposed (B[k][n]=Bm[n][k]).
__device__ __forceinline__ void gemm_bt(const __bf16* A, const __bf16* Bm,
                                        int m0, v8f acc[4], int lane) {
#pragma unroll
  for (int kb = 0; kb < 64; kb += 32) {
    v16bf a = load_a(A, m0, kb, lane);
#pragma unroll
    for (int t = 0; t < 4; ++t) {
      v16bf b = load_bt(Bm, t * 16, kb, lane);
      acc[t] = __builtin_amdgcn_wmma_f32_16x16x32_bf16(
          false, a, false, b, (short)0, acc[t], false, false);
    }
  }
}

__global__ void __launch_bounds__(128, 1)
swin_window_attn(const float* __restrict__ skip, const float* __restrict__ x,
                 const float* __restrict__ pe,
                 const float* __restrict__ Wq, const float* __restrict__ Wk,
                 const float* __restrict__ Wv, const float* __restrict__ Wl,
                 float* __restrict__ out) {
  extern __shared__ char smem_raw[];
  __bf16* Xs  = reinterpret_cast<__bf16*>(smem_raw);   // x window   [64][LDA]
  __bf16* Ssk = Xs  + 64 * LDA;                        // skip window
  __bf16* Qs  = Ssk + 64 * LDA;
  __bf16* Ks  = Qs  + 64 * LDA;
  __bf16* Vs  = Ks  + 64 * LDA;
  __bf16* Ps  = Vs  + 64 * LDA;                        // softmax probs
  __bf16* Os  = Ps  + 64 * LDA;                        // attn output
  __bf16* Wqs = Os  + 64 * LDA;
  __bf16* Wks = Wqs + 64 * LDA;
  __bf16* Wvs = Wks + 64 * LDA;
  __bf16* Wls = Wvs + 64 * LDA;
  float*  Ys  = reinterpret_cast<float*>(Wls + 64 * LDA);  // [64][LDY] f32
  float*  PeS = Ys + 64 * LDY;                             // 15x15 bias table

  const int tid  = threadIdx.x;
  const int lane = tid & 31;
  const int wave = tid >> 5;

  const int win  = blockIdx.x & 255;
  const int head = (blockIdx.x >> 8) & 7;
  const int b    = blockIdx.x >> 11;
  const int nh   = win >> 4;
  const int nw   = win & 15;
  const bool lastRow = (nh == NW - 1);
  const bool lastCol = (nw == NW - 1);

  const size_t plane = (size_t)IMG * IMG;
  const float* xbase = x    + ((size_t)b * CH + head * HD) * plane;
  const float* sbase = skip + ((size_t)b * CH + head * HD) * plane;
  float*       obase = out  + ((size_t)b * CH + head * HD) * plane;

  // ---- stage window of x/skip (rolled by -DISP) into LDS as bf16.
  // One float4 = 4 consecutive tokens of one window row at one channel d.
  // colbase is a multiple of 4 and the roll wraps on quad boundaries, so
  // every quad is a contiguous, 16B-aligned global b128.
#pragma unroll
  for (int it = 0; it < 8; ++it) {
    const int l  = it * 128 + tid;       // [0,1024): {d:64} x {i:8} x {jg:2}
    const int d  = l >> 4;
    const int i  = (l >> 1) & 7;
    const int jg = l & 1;
    const int row     = (nh * WSZ + i + DISP) & (IMG - 1);
    const int colbase = (nw * WSZ + jg * 4 + DISP) & (IMG - 1);
    const size_t g = (size_t)d * plane + (size_t)row * IMG + colbase;
    const float4 xv = *reinterpret_cast<const float4*>(xbase + g);
    const float4 sv = *reinterpret_cast<const float4*>(sbase + g);
    const int t0 = i * WSZ + jg * 4;
    Xs [(t0 + 0) * LDA + d] = (__bf16)xv.x;
    Xs [(t0 + 1) * LDA + d] = (__bf16)xv.y;
    Xs [(t0 + 2) * LDA + d] = (__bf16)xv.z;
    Xs [(t0 + 3) * LDA + d] = (__bf16)xv.w;
    Ssk[(t0 + 0) * LDA + d] = (__bf16)sv.x;
    Ssk[(t0 + 1) * LDA + d] = (__bf16)sv.y;
    Ssk[(t0 + 2) * LDA + d] = (__bf16)sv.z;
    Ssk[(t0 + 3) * LDA + d] = (__bf16)sv.w;
  }
  // ---- stage the four 64x64 weights (row-major, contiguous along d).
#pragma unroll
  for (int it = 0; it < 8; ++it) {
    const int le = (it * 128 + tid) * 4;   // element index, multiple of 4
    const int r = le >> 6, c = le & 63;
    const float4 q4 = *reinterpret_cast<const float4*>(Wq + le);
    const float4 k4 = *reinterpret_cast<const float4*>(Wk + le);
    const float4 v4 = *reinterpret_cast<const float4*>(Wv + le);
    const float4 l4 = *reinterpret_cast<const float4*>(Wl + le);
    __bf16* wq = Wqs + r * LDA + c;  __bf16* wk = Wks + r * LDA + c;
    __bf16* wv = Wvs + r * LDA + c;  __bf16* wl = Wls + r * LDA + c;
    wq[0] = (__bf16)q4.x; wq[1] = (__bf16)q4.y; wq[2] = (__bf16)q4.z; wq[3] = (__bf16)q4.w;
    wk[0] = (__bf16)k4.x; wk[1] = (__bf16)k4.y; wk[2] = (__bf16)k4.z; wk[3] = (__bf16)k4.w;
    wv[0] = (__bf16)v4.x; wv[1] = (__bf16)v4.y; wv[2] = (__bf16)v4.z; wv[3] = (__bf16)v4.w;
    wl[0] = (__bf16)l4.x; wl[1] = (__bf16)l4.y; wl[2] = (__bf16)l4.z; wl[3] = (__bf16)l4.w;
  }
  for (int l = tid; l < 225; l += 128) PeS[l] = pe[l];
  __syncthreads();

  const int m0 = wave * 16;               // this wave's row strip

  // ---- projections: Q = Xw*Wq^T, K = Xw*Wk^T, V = Sw*Wv^T
  { v8f a[4] = {}; gemm_bt(Xs,  Wqs, m0, a, lane); store_strip(Qs, m0, a, lane); }
  { v8f a[4] = {}; gemm_bt(Xs,  Wks, m0, a, lane); store_strip(Ks, m0, a, lane); }
  { v8f a[4] = {}; gemm_bt(Ssk, Wvs, m0, a, lane); store_strip(Vs, m0, a, lane); }
  __syncthreads();                         // all waves need full K, V

  // ---- S = Q K^T  (K^T is just "transposed B" of the token-major K)
  v8f sacc[4] = {};
  gemm_bt(Qs, Ks, m0, sacc, lane);

  // ---- bias + shift mask + softmax (rows live in 16-lane half-groups)
  const int half = lane >> 4, nl = lane & 15;
#pragma unroll
  for (int r = 0; r < 8; ++r) {
    const int qt = m0 + (half << 3) + r;
    const int i1 = qt >> 3, j1 = qt & 7;
    float rowv[4];
    float mx = -3.0e38f;
#pragma unroll
    for (int t = 0; t < 4; ++t) {
      const int kt = t * 16 + nl;
      const int i2 = kt >> 3, j2 = kt & 7;
      float v = sacc[t][r] * 0.125f + PeS[(i2 - i1 + 7) * 15 + (j2 - j1 + 7)];
      const bool dead = (lastRow && ((i1 >= DISP) != (i2 >= DISP))) ||
                        (lastCol && ((j1 >= DISP) != (j2 >= DISP)));
      v = dead ? -3.0e38f : v;
      rowv[t] = v;
      mx = fmaxf(mx, v);
    }
#pragma unroll
    for (int off = 8; off; off >>= 1) mx = fmaxf(mx, __shfl_xor(mx, off, 32));
    float sum = 0.0f;
#pragma unroll
    for (int t = 0; t < 4; ++t) { float e = __expf(rowv[t] - mx); rowv[t] = e; sum += e; }
#pragma unroll
    for (int off = 8; off; off >>= 1) sum += __shfl_xor(sum, off, 32);
    const float inv = 1.0f / sum;
#pragma unroll
    for (int t = 0; t < 4; ++t)
      Ps[qt * LDA + t * 16 + nl] = (__bf16)(rowv[t] * inv);
  }

  // ---- O = P V   (wave-private P strip; V already barriered)
  v8f oacc[4] = {};
#pragma unroll
  for (int kb = 0; kb < 64; kb += 32) {
    v16bf a = load_a(Ps, m0, kb, lane);
#pragma unroll
    for (int t = 0; t < 4; ++t) {
      v16bf bfr = load_bn(Vs, kb, t * 16, lane);
      oacc[t] = __builtin_amdgcn_wmma_f32_16x16x32_bf16(
          false, a, false, bfr, (short)0, oacc[t], false, false);
    }
  }
  store_strip(Os, m0, oacc, lane);

  // ---- Y = O Wl^T
  v8f yacc[4] = {};
  gemm_bt(Os, Wls, m0, yacc, lane);
#pragma unroll
  for (int t = 0; t < 4; ++t)
#pragma unroll
    for (int r = 0; r < 8; ++r)
      Ys[(m0 + (half << 3) + r) * LDY + t * 16 + nl] = yacc[t][r];
  __syncthreads();

  // ---- scatter back to global as float4 (output roll cancels input roll)
#pragma unroll
  for (int it = 0; it < 8; ++it) {
    const int l  = it * 128 + tid;
    const int d  = l >> 4;
    const int i  = (l >> 1) & 7;
    const int jg = l & 1;
    const int row     = (nh * WSZ + i + DISP) & (IMG - 1);
    const int colbase = (nw * WSZ + jg * 4 + DISP) & (IMG - 1);
    const int t0 = i * WSZ + jg * 4;
    float4 yv;
    yv.x = Ys[(t0 + 0) * LDY + d];
    yv.y = Ys[(t0 + 1) * LDY + d];
    yv.z = Ys[(t0 + 2) * LDY + d];
    yv.w = Ys[(t0 + 3) * LDY + d];
    *reinterpret_cast<float4*>(obase + (size_t)d * plane + (size_t)row * IMG + colbase) = yv;
  }
}

extern "C" void kernel_launch(void* const* d_in, const int* in_sizes, int n_in,
                              void* d_out, int out_size, void* d_ws, size_t ws_size,
                              hipStream_t stream) {
  (void)in_sizes; (void)n_in; (void)out_size; (void)d_ws; (void)ws_size;
  const float* skip = (const float*)d_in[0];
  const float* x    = (const float*)d_in[1];
  const float* pe   = (const float*)d_in[2];
  const float* Wq   = (const float*)d_in[3];
  const float* Wk   = (const float*)d_in[4];
  const float* Wv   = (const float*)d_in[5];
  const float* Wl   = (const float*)d_in[6];
  float* out = (float*)d_out;

  constexpr int    nblocks = 8 * 8 * 256;            // batch * heads * windows
  constexpr size_t SMEM = (size_t)11 * 64 * LDA * 2  // 11 bf16 matrices
                        + (size_t)64 * LDY * 4       // f32 Y staging
                        + 225 * 4;                   // pe table
  swin_window_attn<<<nblocks, 128, SMEM, stream>>>(skip, x, pe, Wq, Wk, Wv, Wl, out);
}
